// AudioDALLE_8924942041329
// MI455X (gfx1250) — compile-verified, matrix operand order checked by reading, not run on hardware
//
#include <hip/hip_runtime.h>
#include <math.h>

typedef __attribute__((ext_vector_type(16))) __bf16 v16bf;
typedef __attribute__((ext_vector_type(8)))  float  v8f;
typedef unsigned short ushort_t;

constexpr int BB    = 4;
constexpr int AUDIO = 128;
constexpr int FEAT  = 128;
constexpr int DIM   = 512;
constexpr int HEADS = 8;
constexpr int DH    = 64;
constexpr int G     = 32;
constexpr int IMG   = 1024;
constexpr int VOCAB = 8192;
constexpr int TLEN  = AUDIO + 1;     // 129
constexpr int SEQ   = AUDIO + IMG;   // 1152
constexpr int NP1   = SEQ + 1;       // 1153 (qkv input padded with zero row)
constexpr int M1    = BB * NP1;      // 4612
constexpr int M0    = BB * SEQ;      // 4608
constexpr int MIMG  = BB * IMG;      // 4096
constexpr int QKVW  = 3 * DIM;       // 1536
constexpr float SCALE  = 0.125f;     // DH^-0.5
constexpr float NEGBIG = -1e30f;

// ---------------- small device helpers ----------------
static __device__ __forceinline__ ushort_t f2bf(float f) {
  unsigned u = __float_as_uint(f);
  u += 0x7FFFu + ((u >> 16) & 1u);   // round-to-nearest-even
  return (ushort_t)(u >> 16);
}
static __device__ __forceinline__ float wsum(float v) {
#pragma unroll
  for (int o = 16; o > 0; o >>= 1) v += __shfl_xor(v, o, 32);
  return v;
}
static __device__ __forceinline__ float wmax(float v) {
#pragma unroll
  for (int o = 16; o > 0; o >>= 1) v = fmaxf(v, __shfl_xor(v, o, 32));
  return v;
}

struct Frag32B { uint4 a, b; };   // 16 bf16 = one WMMA A/B fragment per lane

static __device__ __forceinline__ v16bf load_frag(const ushort_t* row, int o0, int o1) {
  Frag32B f;
  f.a = *(const uint4*)(row + o0);
  f.b = *(const uint4*)(row + o1);
  return __builtin_bit_cast(v16bf, f);
}

// ---------------- weight transpose + bf16 convert: [K][N] f32 -> [N][K] bf16 ----------------
__global__ void k_w_transpose(const float* __restrict__ src, ushort_t* __restrict__ dst,
                              int K, int N) {
  int idx = blockIdx.x * blockDim.x + threadIdx.x;
  if (idx >= K * N) return;
  int k = idx / N, n = idx % N;
  dst[(size_t)n * K + k] = f2bf(src[idx]);
}

// ---------------- embeddings ----------------
__global__ void k_embed_audio(const float* __restrict__ audio, const float* __restrict__ Win,
                              const float* __restrict__ bin, const float* __restrict__ apos,
                              float* __restrict__ x) {
  int idx = blockIdx.x * blockDim.x + threadIdx.x;   // BB*AUDIO*DIM
  int j = idx % DIM, row = idx / DIM;
  int i = row % AUDIO, b = row / AUDIO;
  float acc = bin[j] + apos[i * DIM + j];
  const float* ar = audio + (size_t)row * FEAT;
#pragma unroll 4
  for (int k = 0; k < FEAT; k++) acc += ar[k] * Win[k * DIM + j];
  x[((size_t)(b * SEQ + i)) * DIM + j] = acc;
}

__global__ void k_embed_img(const int* __restrict__ image, const float* __restrict__ emb,
                            const float* __restrict__ axr, const float* __restrict__ axc,
                            float* __restrict__ x) {
  int idx = blockIdx.x * blockDim.x + threadIdx.x;   // BB*IMG*DIM
  int j = idx % DIM, row = idx / DIM;
  int t = row % IMG, b = row / IMG;
  int tok = image[b * IMG + t];
  float v = emb[(size_t)tok * DIM + j] + axr[(t >> 5) * DIM + j] + axc[(t & 31) * DIM + j];
  x[((size_t)(b * SEQ + AUDIO + t)) * DIM + j] = v;
}

// ---------------- layernorm (wave per 512-wide row) -> bf16 ----------------
// padded variant: rows laid out [B][NP1], row i==SEQ written as zeros (the _qkv pad row)
__global__ void k_ln_pad(const float* __restrict__ x, const float* __restrict__ sc,
                         const float* __restrict__ bi, ushort_t* __restrict__ out) {
  int w = blockIdx.x * (blockDim.x >> 5) + (threadIdx.x >> 5);
  int lane = threadIdx.x & 31;
  if (w >= M1) return;
  int i = w % NP1, b = w / NP1;
  ushort_t* orow = out + (size_t)w * DIM;
  if (i == SEQ) {
#pragma unroll
    for (int t2 = 0; t2 < 16; t2++) orow[t2 * 32 + lane] = 0;
    return;
  }
  const float* xr = x + (size_t)(b * SEQ + i) * DIM;
  float v[16]; float s = 0.f, sq = 0.f;
#pragma unroll
  for (int t2 = 0; t2 < 16; t2++) { float f = xr[t2 * 32 + lane]; v[t2] = f; s += f; sq += f * f; }
  s = wsum(s); sq = wsum(sq);
  float mean = s * (1.f / DIM);
  float rstd = rsqrtf(sq * (1.f / DIM) - mean * mean + 1e-5f);
#pragma unroll
  for (int t2 = 0; t2 < 16; t2++) {
    int c = t2 * 32 + lane;
    orow[c] = f2bf((v[t2] - mean) * rstd * sc[c] + bi[c]);
  }
}

// generic row-mapped layernorm: xrow = (w/chunk)*bstride + boff + (w%chunk)
__global__ void k_ln_rows(const float* __restrict__ x, const float* __restrict__ sc,
                          const float* __restrict__ bi, ushort_t* __restrict__ out,
                          int rows, int chunk, int bstride, int boff) {
  int w = blockIdx.x * (blockDim.x >> 5) + (threadIdx.x >> 5);
  int lane = threadIdx.x & 31;
  if (w >= rows) return;
  int xrow = (w / chunk) * bstride + boff + (w % chunk);
  const float* xr = x + (size_t)xrow * DIM;
  float v[16]; float s = 0.f, sq = 0.f;
#pragma unroll
  for (int t2 = 0; t2 < 16; t2++) { float f = xr[t2 * 32 + lane]; v[t2] = f; s += f; sq += f * f; }
  s = wsum(s); sq = wsum(sq);
  float mean = s * (1.f / DIM);
  float rstd = rsqrtf(sq * (1.f / DIM) - mean * mean + 1e-5f);
  ushort_t* orow = out + (size_t)w * DIM;
#pragma unroll
  for (int t2 = 0; t2 < 16; t2++) {
    int c = t2 * 32 + lane;
    orow[c] = f2bf((v[t2] - mean) * rstd * sc[c] + bi[c]);
  }
}

// ---------------- WMMA bf16 GEMM: C[M][N] f32 = A[M][K] bf16 * Bt[N][K]^T bf16 ----------------
// 64x64 macro tile per 4-wave block. Each wave computes a 16(M) x 64(N) strip (4 accumulators),
// with the A fragment loaded once per K-step and reused across the 4 N tiles.
// The shared 64x32 B strip is staged to LDS ONCE per block per K-step via the CDNA5
// async-tensor path (GLOBAL_LOAD_ASYNC_TO_LDS_B128, ASYNCcnt) -> 4x less global B traffic;
// waves then read fragments back with ds_load_b128 in WMMA lane striping.
__global__ void k_gemm_bf16(const ushort_t* __restrict__ A, const ushort_t* __restrict__ Bt,
                            float* __restrict__ C, int Mr, int N, int K) {
  __shared__ ushort_t bsh[64 * 32];                        // 4KB: 64 N-rows x 32 k
  int tid  = threadIdx.x;
  int wave = tid >> 5, lane = tid & 31;
  int half = lane >> 4, lm = lane & 15;
  int m0 = blockIdx.y * 64 + wave * 16;
  int n0 = blockIdx.x * 64;
  int am = m0 + lm; if (am > Mr - 1) am = Mr - 1;          // clamp ragged M for loads
  const ushort_t* arow = A + (size_t)am * K;
  // staging assignment: thread covers 16 contiguous elements (32B) of one B row
  int srow = tid >> 1, spart = tid & 1;
  const ushort_t* bsrc = Bt + (size_t)(n0 + srow) * K + spart * 16;
  unsigned bdst = (unsigned)(size_t)(bsh + srow * 32 + spart * 16);   // LDS byte address
  v8f acc[4];
#pragma unroll
  for (int t = 0; t < 4; t++) acc[t] = (v8f){0.f, 0.f, 0.f, 0.f, 0.f, 0.f, 0.f, 0.f};
  for (int k0 = 0; k0 < K; k0 += 32) {
    // async DMA the 64x32 B strip into LDS (IOFFSET applies to both LDS and mem sides)
    unsigned long long ga = (unsigned long long)(size_t)(bsrc + k0);
    asm volatile("global_load_async_to_lds_b128 %0, %1, off"
                 :: "v"(bdst), "v"(ga) : "memory");
    asm volatile("global_load_async_to_lds_b128 %0, %1, off offset:16"
                 :: "v"(bdst), "v"(ga) : "memory");
    int o0 = k0 + half * 8, o1 = k0 + 16 + half * 8;       // ISA 16-bit A lane striping
    v16bf fa = load_frag(arow, o0, o1);                    // overlaps with async copy
    asm volatile("s_wait_asynccnt 0x0" ::: "memory");
    __syncthreads();
#pragma unroll
    for (int t = 0; t < 4; t++) {
      const ushort_t* bl = bsh + (t * 16 + lm) * 32;       // B fragment from LDS
      v16bf fb = load_frag(bl, half * 8, 16 + half * 8);
      acc[t] = __builtin_amdgcn_wmma_f32_16x16x32_bf16(false, fa, false, fb,
                                                       (short)0, acc[t], false, false);
    }
    __syncthreads();                                       // strip consumed; safe to restage
  }
#pragma unroll
  for (int t = 0; t < 4; t++) {
#pragma unroll
    for (int r = 0; r < 8; r++) {                          // D layout: lane half -> M+8
      int mm = m0 + half * 8 + r;
      if (mm < Mr) C[(size_t)mm * N + n0 + t * 16 + lm] = acc[t][r];
    }
  }
}

// ---------------- attention: one wave32 per (b,h,query); lane owns dims d and d+32 ----------------
__global__ void k_text_attn(const float* __restrict__ qkv, ushort_t* __restrict__ attn) {
  int w = blockIdx.x * (blockDim.x >> 5) + (threadIdx.x >> 5);
  int lane = threadIdx.x & 31;
  if (w >= BB * HEADS * TLEN) return;
  int i = w % TLEN, h = (w / TLEN) % HEADS, b = w / (TLEN * HEADS);
  size_t base = (size_t)b * NP1;
  const float* qr = qkv + (base + i) * QKVW + h * DH;
  float q0 = qr[lane] * SCALE, q1 = qr[lane + 32] * SCALE;
  int nk = i + 1;                                          // causal
  float dl[5];
#pragma unroll
  for (int s = 0; s < 5; s++) {
    for (int jj = 0; jj < 32; jj++) {
      int j = s * 32 + jj; if (j >= nk) break;
      const float* kr = qkv + (base + j) * QKVW + DIM + h * DH;
      float p = wsum(q0 * kr[lane] + q1 * kr[lane + 32]);
      if (jj == lane) dl[s] = p;
    }
  }
  float mx = NEGBIG;
#pragma unroll
  for (int s = 0; s < 5; s++) { if (s * 32 + lane < nk) mx = fmaxf(mx, dl[s]); }
  mx = wmax(mx);
  float se = 0.f;
#pragma unroll
  for (int s = 0; s < 5; s++) { if (s * 32 + lane < nk) { dl[s] = __expf(dl[s] - mx); se += dl[s]; } }
  se = wsum(se);
  float inv = 1.f / se, a0 = 0.f, a1 = 0.f;
#pragma unroll
  for (int s = 0; s < 5; s++) {
    for (int jj = 0; jj < 32; jj++) {
      int j = s * 32 + jj; if (j >= nk) break;
      float wgt = __shfl(dl[s], jj, 32) * inv;
      const float* vr = qkv + (base + j) * QKVW + 2 * DIM + h * DH;
      a0 += wgt * vr[lane]; a1 += wgt * vr[lane + 32];
    }
  }
  ushort_t* orow = attn + (base + i) * DIM + h * DH;
  orow[lane] = f2bf(a0); orow[lane + 32] = f2bf(a1);
}

__global__ void k_axial_attn(const float* __restrict__ qkv, ushort_t* __restrict__ attn, int axis) {
  int w = blockIdx.x * (blockDim.x >> 5) + (threadIdx.x >> 5);
  int lane = threadIdx.x & 31;
  if (w >= BB * HEADS * IMG) return;
  int t = w % IMG, h = (w / IMG) % HEADS, b = w / (IMG * HEADS);
  int r = t >> 5, c = t & 31;
  int ii = axis ? r : c;                                   // causal position along axis
  int xx = axis ? c : r;                                   // fixed axis coordinate
  size_t base = (size_t)b * NP1;
  const float* qr = qkv + (base + TLEN + t) * QKVW + h * DH;
  float q0 = qr[lane] * SCALE, q1 = qr[lane + 32] * SCALE;
  int nk = TLEN + ii + 1;                                  // all text keys + causal axial keys
  float dl[6];
#pragma unroll
  for (int s = 0; s < 6; s++) {
    for (int jj = 0; jj < 32; jj++) {
      int j = s * 32 + jj; if (j >= nk) break;
      int seq;
      if (j < TLEN) seq = j;
      else { int jd = j - TLEN; seq = TLEN + (axis ? (jd * G + xx) : (xx * G + jd)); }
      const float* kr = qkv + (base + seq) * QKVW + DIM + h * DH;
      float p = wsum(q0 * kr[lane] + q1 * kr[lane + 32]);
      if (jj == lane) dl[s] = p;
    }
  }
  float mx = NEGBIG;
#pragma unroll
  for (int s = 0; s < 6; s++) { if (s * 32 + lane < nk) mx = fmaxf(mx, dl[s]); }
  mx = wmax(mx);
  float se = 0.f;
#pragma unroll
  for (int s = 0; s < 6; s++) { if (s * 32 + lane < nk) { dl[s] = __expf(dl[s] - mx); se += dl[s]; } }
  se = wsum(se);
  float inv = 1.f / se, a0 = 0.f, a1 = 0.f;
#pragma unroll
  for (int s = 0; s < 6; s++) {
    for (int jj = 0; jj < 32; jj++) {
      int j = s * 32 + jj; if (j >= nk) break;
      float wgt = __shfl(dl[s], jj, 32) * inv;
      int seq;
      if (j < TLEN) seq = j;
      else { int jd = j - TLEN; seq = TLEN + (axis ? (jd * G + xx) : (xx * G + jd)); }
      const float* vr = qkv + (base + seq) * QKVW + 2 * DIM + h * DH;
      a0 += wgt * vr[lane]; a1 += wgt * vr[lane + 32];
    }
  }
  ushort_t* orow = attn + (base + TLEN + t) * DIM + h * DH;
  orow[lane] = f2bf(a0); orow[lane + 32] = f2bf(a1);
}

__global__ void k_conv_attn(const float* __restrict__ qkv, ushort_t* __restrict__ attn) {
  int w = blockIdx.x * (blockDim.x >> 5) + (threadIdx.x >> 5);
  int lane = threadIdx.x & 31;
  if (w >= BB * HEADS * IMG) return;
  int t = w % IMG, h = (w / IMG) % HEADS, b = w / (IMG * HEADS);
  int r = t >> 5, c = t & 31;
  size_t base = (size_t)b * NP1;
  const float* qr = qkv + (base + TLEN + t) * QKVW + h * DH;
  float q0 = qr[lane] * SCALE, q1 = qr[lane + 32] * SCALE;
  const int nk = TLEN + 25;                                // 129 text + 5x5 neighborhood
  float dl[5];
#pragma unroll
  for (int s = 0; s < 5; s++) {
    for (int jj = 0; jj < 32; jj++) {
      int j = s * 32 + jj; if (j >= nk) break;
      float p;
      if (j < TLEN) {
        const float* kr = qkv + (base + j) * QKVW + DIM + h * DH;
        p = wsum(q0 * kr[lane] + q1 * kr[lane + 32]);
      } else {
        int sd = j - TLEN;
        int rr = r + sd / 5 - 2, cc = c + sd % 5 - 2;
        bool valid = ((unsigned)rr < (unsigned)G) && ((unsigned)cc < (unsigned)G) &&
                     (rr * G + cc) <= t;                   // OOB or future -> masked
        if (valid) {
          const float* kr = qkv + (base + TLEN + rr * G + cc) * QKVW + DIM + h * DH;
          p = wsum(q0 * kr[lane] + q1 * kr[lane + 32]);
        } else p = NEGBIG;
      }
      if (jj == lane) dl[s] = p;
    }
  }
  float mx = NEGBIG;
#pragma unroll
  for (int s = 0; s < 5; s++) { if (s * 32 + lane < nk) mx = fmaxf(mx, dl[s]); }
  mx = wmax(mx);
  float se = 0.f;
#pragma unroll
  for (int s = 0; s < 5; s++) { if (s * 32 + lane < nk) { dl[s] = __expf(dl[s] - mx); se += dl[s]; } }
  se = wsum(se);
  float inv = 1.f / se, a0 = 0.f, a1 = 0.f;
#pragma unroll
  for (int s = 0; s < 5; s++) {
    for (int jj = 0; jj < 32; jj++) {
      int j = s * 32 + jj; if (j >= nk) break;
      int seq;
      if (j < TLEN) seq = j;
      else {
        int sd = j - TLEN;
        int rr = r + sd / 5 - 2, cc = c + sd % 5 - 2;
        bool valid = ((unsigned)rr < (unsigned)G) && ((unsigned)cc < (unsigned)G) &&
                     (rr * G + cc) <= t;
        if (!valid) continue;                              // weight ~0 anyway
        seq = TLEN + rr * G + cc;
      }
      float wgt = __shfl(dl[s], jj, 32) * inv;
      const float* vr = qkv + (base + seq) * QKVW + 2 * DIM + h * DH;
      a0 += wgt * vr[lane]; a1 += wgt * vr[lane + 32];
    }
  }
  ushort_t* orow = attn + (base + TLEN + t) * DIM + h * DH;
  orow[lane] = f2bf(a0); orow[lane + 32] = f2bf(a1);
}

// ---------------- residual epilogues / GEGLU ----------------
__global__ void k_resid1(float* __restrict__ x, const float* __restrict__ proj,
                         const float* __restrict__ bo, const float* __restrict__ g1) {
  int idx = blockIdx.x * blockDim.x + threadIdx.x;   // M0*DIM
  int j = idx % DIM, w = idx / DIM;
  int i = w % SEQ, b = w / SEQ;
  x[idx] += (proj[(size_t)(b * NP1 + i) * DIM + j] + bo[j]) * g1[j];
}

__global__ void k_resid2(float* __restrict__ x, const float* __restrict__ f,
                         const float* __restrict__ b2, const float* __restrict__ g2) {
  int idx = blockIdx.x * blockDim.x + threadIdx.x;   // M0*DIM
  int j = idx % DIM;
  x[idx] += (f[idx] + b2[j]) * g2[j];
}

__global__ void k_glu(const float* __restrict__ ug, const float* __restrict__ b1,
                      ushort_t* __restrict__ gg) {
  int idx = blockIdx.x * blockDim.x + threadIdx.x;   // M0*(4*DIM)
  int j = idx % (4 * DIM), row = idx / (4 * DIM);
  float u = ug[(size_t)row * (8 * DIM) + j] + b1[j];
  float g = ug[(size_t)row * (8 * DIM) + 4 * DIM + j] + b1[4 * DIM + j];
  float ge = 0.5f * g * (1.f + erff(g * 0.70710678118654752f));   // exact gelu
  gg[idx] = f2bf(u * ge);
}

// ---------------- fused logits GEMM + online log-sum-exp + NLL ----------------
// one wave per 16-row tile; sweeps VOCAB in 16x64 register-blocked WMMA strips,
// staging each strip to LDS for the online softmax update; never materializes logits.
__global__ void k_logits_loss(const ushort_t* __restrict__ A, const ushort_t* __restrict__ Bt,
                              const float* __restrict__ bias, const int* __restrict__ image,
                              float* __restrict__ rowloss) {
  __shared__ float tile_s[16][65];
  __shared__ float mrun[16], srun[16], tgt[16];
  int lane = threadIdx.x & 31;
  int half = lane >> 4, lm = lane & 15;
  int m0 = blockIdx.x * 16;
  if (lane < 16) { mrun[lane] = NEGBIG; srun[lane] = 0.f; tgt[lane] = 0.f; }
  __syncthreads();
  const ushort_t* arow = A + (size_t)(m0 + lm) * DIM;
  for (int nb = 0; nb < VOCAB / 64; nb++) {
    const ushort_t* brow[4];
#pragma unroll
    for (int t = 0; t < 4; t++) brow[t] = Bt + (size_t)(nb * 64 + t * 16 + lm) * DIM;
    v8f acc[4];
#pragma unroll
    for (int t = 0; t < 4; t++) acc[t] = (v8f){0.f, 0.f, 0.f, 0.f, 0.f, 0.f, 0.f, 0.f};
    for (int k0 = 0; k0 < DIM; k0 += 32) {
      int o0 = k0 + half * 8, o1 = k0 + 16 + half * 8;
      v16bf fa = load_frag(arow, o0, o1);
#pragma unroll
      for (int t = 0; t < 4; t++) {
        v16bf fb = load_frag(brow[t], o0, o1);
        acc[t] = __builtin_amdgcn_wmma_f32_16x16x32_bf16(false, fa, false, fb,
                                                         (short)0, acc[t], false, false);
      }
    }
#pragma unroll
    for (int t = 0; t < 4; t++) {
      float bv = bias[nb * 64 + t * 16 + lm];
#pragma unroll
      for (int rr = 0; rr < 8; rr++) tile_s[half * 8 + rr][t * 16 + lm] = acc[t][rr] + bv;
    }
    __syncthreads();
    if (lane < 16) {
      int rid = m0 + lane;
      int tcol = image[(rid >> 10) * IMG + (rid & (IMG - 1))];
      float mold = mrun[lane], mx = mold;
#pragma unroll 8
      for (int q = 0; q < 64; q++) mx = fmaxf(mx, tile_s[lane][q]);
      float s = srun[lane] * __expf(mold - mx);
#pragma unroll 8
      for (int q = 0; q < 64; q++) s += __expf(tile_s[lane][q] - mx);
      mrun[lane] = mx; srun[lane] = s;
      if ((tcol >> 6) == nb) tgt[lane] = tile_s[lane][tcol & 63];
    }
    __syncthreads();
  }
  if (lane < 16) rowloss[m0 + lane] = mrun[lane] + __logf(srun[lane]) - tgt[lane];
}

__global__ void k_loss_reduce(const float* __restrict__ rl, float* __restrict__ out) {
  __shared__ float sh[256];
  int t = threadIdx.x;
  float s = 0.f;
  for (int i = t; i < MIMG; i += 256) s += rl[i];
  sh[t] = s; __syncthreads();
  for (int k = 128; k > 0; k >>= 1) { if (t < k) sh[t] += sh[t + k]; __syncthreads(); }
  if (t == 0) out[0] = sh[0] / (float)MIMG;
}

// ---------------- host driver ----------------
extern "C" void kernel_launch(void* const* d_in, const int* in_sizes, int n_in,
                              void* d_out, int out_size, void* d_ws, size_t ws_size,
                              hipStream_t stream) {
  (void)in_sizes; (void)n_in; (void)out_size; (void)ws_size;
  const float* audio = (const float*)d_in[0];
  const int*   image = (const int*)d_in[1];
  int pi = 2;
  auto F = [&]() -> const float* { return (const float*)d_in[pi++]; };
  const float* img_emb   = F();
  const float* audio_pos = F();
  const float* ax_row    = F();
  const float* ax_col    = F();
  const float* W_in      = F();
  const float* b_in      = F();
  const float* ln_out_s  = F();
  const float* ln_out_b  = F();
  const float* W_logit   = F();
  const float* b_logit   = F();
  struct LP { const float *ln1_s,*ln1_b,*Wqkv,*Wo,*bo,*g1,*ln2_s,*ln2_b,*W1,*b1,*W2,*b2,*g2; } lp[4];
  for (int l = 0; l < 4; l++) {
    lp[l].ln1_s = F(); lp[l].ln1_b = F(); lp[l].Wqkv = F(); lp[l].Wo = F(); lp[l].bo = F();
    lp[l].g1 = F();    lp[l].ln2_s = F(); lp[l].ln2_b = F(); lp[l].W1 = F(); lp[l].b1 = F();
    lp[l].W2 = F();    lp[l].b2 = F();    lp[l].g2 = F();
  }

  // workspace carve (256B aligned)
  char* base = (char*)d_ws;
  size_t off = 0;
  auto bump = [&](size_t bytes) -> char* {
    char* p = base + off;
    off = (off + bytes + 255) & ~(size_t)255;
    return p;
  };
  float*    xres  = (float*)bump((size_t)M0 * DIM * 4);
  ushort_t* hb    = (ushort_t*)bump((size_t)M1 * DIM * 2);
  ushort_t* h2b   = (ushort_t*)bump((size_t)M0 * DIM * 2);
  ushort_t* loutb = (ushort_t*)bump((size_t)MIMG * DIM * 2);
  float*    rloss = (float*)bump((size_t)MIMG * 4);
  ushort_t *WqkvT[4], *WoT[4], *W1T[4], *W2T[4];
  for (int l = 0; l < 4; l++) {
    WqkvT[l] = (ushort_t*)bump((size_t)DIM * QKVW * 2);
    WoT[l]   = (ushort_t*)bump((size_t)DIM * DIM * 2);
    W1T[l]   = (ushort_t*)bump((size_t)DIM * (8 * DIM) * 2);
    W2T[l]   = (ushort_t*)bump((size_t)(4 * DIM) * DIM * 2);
  }
  ushort_t* WlogT = (ushort_t*)bump((size_t)DIM * VOCAB * 2);
  // phase-overlapped arena: {qkv, attn, proj} for attention phase; {ug, gg, ffo} for FFN phase
  size_t qkvA = ((size_t)M1 * QKVW * 4 + 255) & ~(size_t)255;
  size_t attA = ((size_t)M1 * DIM  * 2 + 255) & ~(size_t)255;
  size_t ugA  = ((size_t)M0 * (8 * DIM) * 4 + 255) & ~(size_t)255;
  size_t ggA  = ((size_t)M0 * (4 * DIM) * 2 + 255) & ~(size_t)255;
  size_t phase1 = qkvA + attA + (size_t)M1 * DIM * 4;
  size_t phase2 = ugA + ggA + (size_t)M0 * DIM * 4;
  char* arena = bump(phase1 > phase2 ? phase1 : phase2);
  float*    qkv   = (float*)arena;
  ushort_t* attnb = (ushort_t*)(arena + qkvA);
  float*    proj  = (float*)(arena + qkvA + attA);
  float*    ug    = (float*)arena;
  ushort_t* ggb   = (ushort_t*)(arena + ugA);
  float*    ffo   = (float*)(arena + ugA + ggA);

  auto gemm = [&](const ushort_t* Aa, const ushort_t* Bb, float* Cc, int Mr, int N, int K) {
    dim3 grid(N / 64, (Mr + 63) / 64);   // 64x64 macro tile per block (4 waves)
    k_gemm_bf16<<<grid, 128, 0, stream>>>(Aa, Bb, Cc, Mr, N, K);
  };
  auto tr = [&](const float* s, ushort_t* d, int K, int N) {
    k_w_transpose<<<((K * N) + 255) / 256, 256, 0, stream>>>(s, d, K, N);
  };

  for (int l = 0; l < 4; l++) {
    tr(lp[l].Wqkv, WqkvT[l], DIM, QKVW);
    tr(lp[l].Wo,   WoT[l],   DIM, DIM);
    tr(lp[l].W1,   W1T[l],   DIM, 8 * DIM);
    tr(lp[l].W2,   W2T[l],   4 * DIM, DIM);
  }
  tr(W_logit, WlogT, DIM, VOCAB);

  k_embed_audio<<<(BB * AUDIO * DIM) / 256, 256, 0, stream>>>(audio, W_in, b_in, audio_pos, xres);
  k_embed_img<<<(BB * IMG * DIM) / 256, 256, 0, stream>>>(image, img_emb, ax_row, ax_col, xres);

  for (int l = 0; l < 4; l++) {
    k_ln_pad<<<(M1 + 3) / 4, 128, 0, stream>>>(xres, lp[l].ln1_s, lp[l].ln1_b, hb);
    gemm(hb, WqkvT[l], qkv, M1, QKVW, DIM);
    k_text_attn<<<(BB * HEADS * TLEN + 3) / 4, 128, 0, stream>>>(qkv, attnb);
    if (l == 3)
      k_conv_attn<<<(BB * HEADS * IMG) / 4, 128, 0, stream>>>(qkv, attnb);
    else
      k_axial_attn<<<(BB * HEADS * IMG) / 4, 128, 0, stream>>>(qkv, attnb, (l == 2) ? 1 : 0);
    gemm(attnb, WoT[l], proj, M1, DIM, DIM);
    k_resid1<<<(M0 * DIM) / 256, 256, 0, stream>>>(xres, proj, lp[l].bo, lp[l].g1);
    k_ln_rows<<<(M0 + 3) / 4, 128, 0, stream>>>(xres, lp[l].ln2_s, lp[l].ln2_b, h2b, M0, M0, 0, 0);
    gemm(h2b, W1T[l], ug, M0, 8 * DIM, DIM);
    k_glu<<<(M0 * 4 * DIM) / 256, 256, 0, stream>>>(ug, lp[l].b1, ggb);
    gemm(ggb, W2T[l], ffo, M0, DIM, 4 * DIM);
    k_resid2<<<(M0 * DIM) / 256, 256, 0, stream>>>(xres, ffo, lp[l].b2, lp[l].g2);
  }

  k_ln_rows<<<(MIMG + 3) / 4, 128, 0, stream>>>(xres, ln_out_s, ln_out_b, loutb, MIMG, IMG, SEQ, AUDIO);
  k_logits_loss<<<MIMG / 16, 32, 0, stream>>>(loutb, WlogT, b_logit, image, rloss);
  k_loss_reduce<<<1, 256, 0, stream>>>(rloss, (float*)d_out);
}
